// ProtoLayer_72997264163167
// MI455X (gfx1250) — matrix operand order repaired and checked by reading, not compiled, not run on hardware
//
#include <hip/hip_runtime.h>
#include <hip/hip_bf16.h>
#include <math.h>

// Problem constants (fixed by setup_inputs)
#define WAY   5
#define SHOT  5
#define WQ    75      // way * qnum = 5*15
#define C     640
#define NPAIR 204480  // C*(C-1)/2
#define NJOB  320     // balanced triangle jobs (row a paired with row 638-a)
#define GSEG  16      // job groups for diff kernel (20 jobs each)
#define ITILE 40      // i-tiles of 16 for kendall kernel (40*16 = 640)
#define MTILE 5       // query tiles of 16 (5*16 = 80 >= 75)

typedef __attribute__((ext_vector_type(16))) _Float16 v16h;
typedef __attribute__((ext_vector_type(8)))  float    v8f;

#if __has_builtin(__builtin_amdgcn_tanhf)
#define FAST_TANH(x) __builtin_amdgcn_tanhf(x)
#else
#define FAST_TANH(x) tanhf(x)
#endif

// sign(x) in {-1,0,+1}: overflow saturates to +/-inf -> med3 clamps to +/-1; 0 -> 0.
__device__ __forceinline__ float fsign(float x) {
    return __builtin_amdgcn_fmed3f(x * 3.0e38f, -1.0f, 1.0f);
}

// ---------------------------------------------------------------------------
// Kernel 1: prototypes = mean over shots.  support [25][640] -> proto [5][640]
// ---------------------------------------------------------------------------
__global__ void proto_kernel(const float* __restrict__ support,
                             float* __restrict__ proto) {
    int idx = blockIdx.x * 256 + threadIdx.x;
    if (idx < WAY * C) {
        int w = idx / C, c = idx % C;
        float s = 0.f;
#pragma unroll
        for (int sh = 0; sh < SHOT; ++sh)
            s += support[(w * SHOT + sh) * C + c];
        proto[idx] = s * (1.0f / SHOT);
    }
}

// ---------------------------------------------------------------------------
// Kernel 2: diffkendall tanh partial sums over the strict upper triangle.
// Grid: (GSEG job-groups, WQ queries), 256 threads.
// Row pairing (a, 638-a): combined length 640 per job.  i-side values are
// hoisted out of the j loop (6 LDS loads per 5 tanh).
// ---------------------------------------------------------------------------
__global__ void diff_kernel(const float* __restrict__ Q,
                            const float* __restrict__ proto,
                            float* __restrict__ diffpart) {
    __shared__ float Qs[C];
    __shared__ float Ps[WAY * C];
    __shared__ float red[256];

    const int q   = blockIdx.y;
    const int g   = blockIdx.x;
    const int tid = threadIdx.x;

    for (int c = tid; c < C; c += 256)        Qs[c] = Q[q * C + c];
    for (int c = tid; c < WAY * C; c += 256)  Ps[c] = proto[c];
    __syncthreads();

    float acc[WAY] = {0.f, 0.f, 0.f, 0.f, 0.f};

    for (int a = g * 20; a < g * 20 + 20; ++a) {
        const int nrow = (a == NJOB - 1) ? 1 : 2;   // job 319 pairs with itself
        for (int rsel = 0; rsel < nrow; ++rsel) {
            const int i = (rsel == 0) ? a : (638 - a);
            const float qi = Qs[i];
            float pi[WAY];
#pragma unroll
            for (int w = 0; w < WAY; ++w) pi[w] = Ps[w * C + i];

            for (int j = i + 1 + tid; j < C; j += 256) {
                const float qp = Qs[j] - qi;
#pragma unroll
                for (int w = 0; w < WAY; ++w) {
                    const float sp = Ps[w * C + j] - pi[w];
                    acc[w] += FAST_TANH(0.5f * qp * sp);
                }
            }
        }
    }

    // deterministic in-block tree reduction, one scalar per way
#pragma unroll
    for (int w = 0; w < WAY; ++w) {
        red[tid] = acc[w];
        __syncthreads();
        for (int s = 128; s > 0; s >>= 1) {
            if (tid < s) red[tid] += red[tid + s];
            __syncthreads();
        }
        if (tid == 0) diffpart[(q * GSEG + g) * WAY + w] = red[0];
        __syncthreads();
    }
}

// ---------------------------------------------------------------------------
// Kernel 3: Kendall sign-matmul via v_wmma_f32_16x16x32_f16.
// Full 640x640 ordered (i,j) grid == 2x triangle (diagonal contributes 0).
// One wave per workgroup; grid (ITILE, MTILE).
//   A[m][k] = sign(Q[q=m][j0+k] - Q[q=m][i])   (16x32 f16, wave32 layout)
//   B[k][n] = sign(S[w=n][j0+k] - S[w=n][i])   (32x16 f16, mirrored layout)
// Branch-free: Ss has a 6th all-zero row for columns n>=5; j-block values are
// loaded once per chunk as float4 and reused across all 16 i values.
// ---------------------------------------------------------------------------
__global__ void kendall_wmma_kernel(const float* __restrict__ Q,
                                    const float* __restrict__ proto,
                                    float* __restrict__ kpart) {
    __shared__ __align__(16) float Qs[16 * C];        // 40 KB
    __shared__ __align__(16) float Ss[(WAY + 1) * C]; // 15 KB, row 5 = zeros

    const int itile = blockIdx.x;   // i in [itile*16, itile*16+16)
    const int mtile = blockIdx.y;   // queries [mtile*16, mtile*16+16)
    const int tid   = threadIdx.x;  // 0..31, one wave

    for (int idx = tid; idx < 16 * C; idx += 32) {
        const int r = idx / C, c = idx % C;
        const int q = mtile * 16 + r;
        Qs[idx] = (q < WQ) ? Q[q * C + c] : 0.f;
    }
    for (int idx = tid; idx < (WAY + 1) * C; idx += 32)
        Ss[idx] = (idx < WAY * C) ? proto[idx] : 0.f;
    __syncthreads();

    const int  lane = tid;
    const bool hi   = lane >= 16;
    const int  row  = lane & 15;                    // A row / B column
    const int  srow = (row < WAY) ? row : WAY;      // zero row for n>=5

    // Preload the 16 i-column values into registers.
    float qi[16], si[16];
#pragma unroll
    for (int t = 0; t < 16; ++t) {
        qi[t] = Qs[row  * C + itile * 16 + t];
        si[t] = Ss[srow * C + itile * 16 + t];
    }

    v8f acc = {};

    for (int jc = 0; jc < C / 32; ++jc) {
        const int j0 = jc * 32;

        // A j-values: K runs {base..base+7} and {base+16..base+23}, base = hi?8:0
        float qv[16], sv[16];
        {
            const float4* qa = (const float4*)&Qs[row * C + j0 + (hi ? 8 : 0)];
            *(float4*)&qv[0]  = qa[0];
            *(float4*)&qv[4]  = qa[1];
            *(float4*)&qv[8]  = qa[4];   // +16 floats
            *(float4*)&qv[12] = qa[5];
            // B j-values: same K runs as A, base = hi?8:0
            const float4* ba = (const float4*)&Ss[srow * C + j0 + (hi ? 8 : 0)];
            *(float4*)&sv[0]  = ba[0];
            *(float4*)&sv[4]  = ba[1];
            *(float4*)&sv[8]  = ba[4];   // +16 floats
            *(float4*)&sv[12] = ba[5];
        }

#pragma unroll 1
        for (int t = 0; t < 16; ++t) {              // 16 i values share qv/sv
            v16h a, b;
#pragma unroll
            for (int h = 0; h < 16; ++h) {
                a[h] = (_Float16)fsign(qv[h] - qi[t]);
                b[h] = (_Float16)fsign(sv[h] - si[t]);
            }
            acc = __builtin_amdgcn_wmma_f32_16x16x32_f16(
                false, a, false, b, (short)0, acc, false, false);
        }
    }

    // D layout: VGPR r -> (M = r + 8*hi, N = lane&15)
    if (row < WAY) {
#pragma unroll
        for (int r = 0; r < 8; ++r) {
            const int m = r + (hi ? 8 : 0);
            const int q = mtile * 16 + m;
            if (q < WQ)
                kpart[(q * WAY + row) * ITILE + itile] = acc[r];
        }
    }
}

// ---------------------------------------------------------------------------
// Kernel 4: deterministic final reduction -> d_out [diff 375 | kendall 375]
// ---------------------------------------------------------------------------
__global__ void reduce_kernel(const float* __restrict__ diffpart,
                              const float* __restrict__ kpart,
                              float* __restrict__ out) {
    const int id = blockIdx.x * 256 + threadIdx.x;
    if (id < WQ * WAY) {
        const int q = id / WAY, w = id % WAY;
        float ds = 0.f;
#pragma unroll
        for (int g = 0; g < GSEG; ++g)
            ds += diffpart[(q * GSEG + g) * WAY + w];
        out[id] = ds * (1.0f / ((float)NPAIR * 0.0125f));   // mean / TEMP

        float ks = 0.f;
        for (int it = 0; it < ITILE; ++it)
            ks += kpart[(q * WAY + w) * ITILE + it];
        out[WQ * WAY + id] = ks * (1.0f / (2.0f * (float)NPAIR)); // full grid / 2 / P
    }
}

// ---------------------------------------------------------------------------
extern "C" void kernel_launch(void* const* d_in, const int* in_sizes, int n_in,
                              void* d_out, int out_size, void* d_ws, size_t ws_size,
                              hipStream_t stream) {
    const float* Q = (const float*)d_in[0];   // [1][75][640]
    const float* S = (const float*)d_in[1];   // [1][25][640]
    float* out = (float*)d_out;               // 750 floats: diff then kendall
    float* ws  = (float*)d_ws;

    float* proto    = ws;                 // 3200 floats
    float* diffpart = ws + 3200;          // 75*16*5   = 6000 floats
    float* kpart    = ws + 9200;          // 75*5*40   = 15000 floats (96.8 KB total)

    proto_kernel<<<(WAY * C + 255) / 256, 256, 0, stream>>>(S, proto);
    diff_kernel<<<dim3(GSEG, WQ), 256, 0, stream>>>(Q, proto, diffpart);
    kendall_wmma_kernel<<<dim3(ITILE, MTILE), 32, 0, stream>>>(Q, proto, kpart);
    reduce_kernel<<<2, 256, 0, stream>>>(diffpart, kpart, out);
}
